// EfficientAdditiveAttention_46557445488980
// MI455X (gfx1250) — matrix operand order repaired and checked by reading, not compiled, not run on hardware
//
#include <hip/hip_runtime.h>
#include <hip/hip_bf16.h>
#include <math.h>

// ---------------------------------------------------------------------------
// EfficientAdditiveAttention, fused for MI455X (gfx1250, wave32, WMMA + TDM).
//   B=8, L=512, D=128.
//   Stage 1 (proj3):      Q/K/V projections via v_wmma_f32_16x16x32_bf16.
//   Stage 2 (attn_fused): TDM double-buffered K streaming into LDS,
//                         f32 tanh-score + softmax, attn@V via WMMA.
// Workspace layout (floats): Qp | Kp | Vp, each B*L*D = 524288 (6 MB total).
// ---------------------------------------------------------------------------

#define Bc 8
#define Lc 512
#define Dc 128
#define BLD (Bc * Lc * Dc)

typedef __attribute__((ext_vector_type(16))) __bf16    v16bf;
typedef __attribute__((ext_vector_type(8)))  float     v8f;
typedef __attribute__((ext_vector_type(4)))  unsigned  v4u;
typedef __attribute__((ext_vector_type(8)))  int       v8i;
typedef __attribute__((ext_vector_type(4)))  int       v4i;

// --------------------------- Stage 1: projections ---------------------------
// grid = (B*L/16, 3), block = 256 (8 waves). Wave w owns output columns
// [16w, 16w+16). Each block covers 16 output rows (one M tile).
__global__ __launch_bounds__(256) void proj3_kernel(
    const float* __restrict__ q, const float* __restrict__ k, const float* __restrict__ v,
    const float* __restrict__ Wq, const float* __restrict__ bq,
    const float* __restrict__ Wk, const float* __restrict__ bk,
    const float* __restrict__ Wv, const float* __restrict__ bv,
    float* __restrict__ Qp, float* __restrict__ Kp, float* __restrict__ Vp)
{
    const float* X;  const float* W;  const float* bias;  float* Y;
    if (blockIdx.y == 0)      { X = q; W = Wq; bias = bq; Y = Qp; }
    else if (blockIdx.y == 1) { X = k; W = Wk; bias = bk; Y = Kp; }
    else                      { X = v; W = Wv; bias = bv; Y = Vp; }

    const int lane = threadIdx.x & 31;
    const int wave = threadIdx.x >> 5;
    const int half = lane >> 4;      // lanes 16-31 hold the "upper K" halves
    const int lr   = lane & 15;

    const int m = blockIdx.x * 16 + lr;     // output row (A-matrix M = lr both halves)
    const int n = wave * 16 + lr;           // output col (B/C/D N = lr both halves)

    const float* __restrict__ xrow = X + (size_t)m * Dc;
    const float* __restrict__ wrow = W + (size_t)n * Dc;   // y = x @ W.T -> B[k][n] = W[n][k]

    v8f c = {};
    #pragma unroll
    for (int k0 = 0; k0 < Dc; k0 += 32) {
        v16bf a, bb;
        // A 16x32 bf16 layout: half 0 -> K = {k0+0..7, k0+16..23}
        //                      half 1 -> K = {k0+8..15, k0+24..31}
        const int ka = k0 + half * 8;
        #pragma unroll
        for (int i = 0; i < 8; ++i) {
            a[i]     = (__bf16)xrow[ka + i];
            a[8 + i] = (__bf16)xrow[ka + 16 + i];
        }
        // B 32x16 bf16 layout: same K split as A (half h -> K = {k0+8h..+7, k0+16+8h..+7})
        const int kb = k0 + half * 8;
        #pragma unroll
        for (int i = 0; i < 8; ++i) {
            bb[i]     = (__bf16)wrow[kb + i];
            bb[8 + i] = (__bf16)wrow[kb + 16 + i];
        }

        c = __builtin_amdgcn_wmma_f32_16x16x32_bf16(
                /*neg_a=*/false, a, /*neg_b=*/false, bb,
                /*c_mod=*/(short)0, c, /*reuse_a=*/false, /*reuse_b=*/false);
    }

    const float bn = bias[n];
    // C/D layout: VGPR r -> row M = half*8 + r, col N = lr
    #pragma unroll
    for (int r = 0; r < 8; ++r)
        Y[(size_t)(blockIdx.x * 16 + half * 8 + r) * Dc + n] = c[r] + bn;
}

// --------------------------- TDM 2D tile load helper ------------------------
// Loads a (rows x 128) f32 tile (row stride 128 floats in memory) into LDS,
// with TDM padding of 4 DWORDs after every 128 DWORDs -> LDS row stride 132
// floats (bank-conflict-free across the 64 banks).
// D# packing per CDNA5 ISA 8.3/8.4:
//   g0: count=1 | lds_addr | global_addr[56:0] | type=2
//   g1: data_size=4B, pad_enable, pad_interval=128DW (code 6), pad_amount=4DW
//       (code 3), tensor_dim0=128, tensor_dim1=512, tile_dim0=128,
//       tile_dim1=rows, tensor_dim0_stride=128
__device__ __forceinline__ void tdm_load_tile(unsigned lds_addr,
                                              const float* __restrict__ gptr,
                                              int rows)
{
    const unsigned long long ga = (unsigned long long)(uintptr_t)gptr;
    v4u g0;
    g0.x = 1u;                                               // count=1 (valid), user mode
    g0.y = lds_addr;                                         // LDS byte address
    g0.z = (unsigned)(ga & 0xFFFFFFFFu);                     // global_addr[31:0]
    g0.w = ((unsigned)(ga >> 32) & 0x01FFFFFFu) | 0x80000000u; // [56:32] | type=2

    v8i g1;
    g1[0] = (int)((2u << 16) | (1u << 20) | (6u << 22) | (3u << 25));
    g1[1] = (int)(128u << 16);        // tensor_dim0 = 128 (low half in [63:48])
    g1[2] = (int)(512u << 16);        // tensor_dim1 = 512 (low half in [95:80])
    g1[3] = (int)(128u << 16);        // tile_dim0 = 128 ([127:112])
    g1[4] = rows;                     // tile_dim1 ([143:128])
    g1[5] = 128;                      // tensor_dim0_stride ([191:160])
    g1[6] = 0;
    g1[7] = 0;

    const v4i z4 = {0, 0, 0, 0};
#if defined(__clang_major__) && (__clang_major__ >= 23)
    const v8i z8 = {0, 0, 0, 0, 0, 0, 0, 0};
    __builtin_amdgcn_tensor_load_to_lds(g0, g1, z4, z4, z8, 0);
#else
    __builtin_amdgcn_tensor_load_to_lds(g0, g1, z4, z4, 0);
#endif
}

// --------------------------- Stage 2: fused attention -----------------------
// grid = B*L/16, block = 256 (8 waves). One 16-row q tile per block.
#define SQK_STRIDE 132   // 128 + 4 (TDM pad) -> distinct banks across rows
#define SS_STRIDE  516

__global__ __launch_bounds__(256) void attn_fused_kernel(
    const float* __restrict__ Qp, const float* __restrict__ Kp, const float* __restrict__ Vp,
    const float* __restrict__ w_delta, const float* __restrict__ b_delta,
    const float* __restrict__ w_sigma, const float* __restrict__ b_sigma,
    const float* __restrict__ w_theta, const float* __restrict__ b_theta,
    float* __restrict__ out)
{
    __shared__ float sQ[16][SQK_STRIDE];
    __shared__ float sK2[2][16][SQK_STRIDE];  // TDM double buffer for K chunks
    __shared__ float sS[16][SS_STRIDE];       // scores -> exp(scores)
    __shared__ float sW[Dc];
    __shared__ float sInv[16];

    const int tid = threadIdx.x;
    const int bx  = blockIdx.x;
    const int b   = bx >> 5;                  // L/16 = 32 q tiles per batch
    const int bL  = b * Lc;

    const float* __restrict__ kbase = Kp + (size_t)bL * Dc;

    // Wave 0 kicks off TDM: Q tile, then K chunk 0 (TDM is in-order per wave,
    // so waiting on the K chunk also covers the Q tile).
    if (tid < 32) {
        tdm_load_tile((unsigned)(uintptr_t)&sQ[0][0], Qp + (size_t)bx * 16 * Dc, 16);
        tdm_load_tile((unsigned)(uintptr_t)&sK2[0][0][0], kbase, 16);
    }

    // Warm L2 for the V tile this block will consume (gfx1250 global_prefetch_b8).
    __builtin_prefetch(Vp + (size_t)(bL + tid) * Dc, 0, 1);
    __builtin_prefetch(Vp + (size_t)(bL + 256 + tid) * Dc, 0, 1);

    // w_all = 1/sqrt(D) + w_delta + w_sigma + w_theta  (base head fused in)
    if (tid < Dc)
        sW[tid] = 0.088388347648318447f + w_delta[tid] + w_sigma[tid] + w_theta[tid];
    const float b_all = b_delta[0] + b_sigma[0] + b_theta[0];

    // ---- score pass: each thread owns one (q, k) pair per 16-k chunk ----
    const int qi = tid & 15;
    const int kl = tid >> 4;
    for (int kc = 0, ib = 0; kc < Lc; kc += 16, ib ^= 1) {
        if (kc + 16 < Lc) {
            if (tid < 32)   // issue next chunk into the other buffer
                tdm_load_tile((unsigned)(uintptr_t)&sK2[ib ^ 1][0][0],
                              kbase + (size_t)(kc + 16) * Dc, 16);
            __builtin_amdgcn_s_wait_tensorcnt(1);   // oldest (current chunk) done
        } else {
            __builtin_amdgcn_s_wait_tensorcnt(0);   // drain
        }
        __syncthreads();

        float acc = 0.f;
        #pragma unroll 8
        for (int d = 0; d < Dc; ++d)
            acc = fmaf(tanhf(sQ[qi][d] + sK2[ib][kl][d]), sW[d], acc);
        sS[qi][kc + kl] = acc + b_all;
        __syncthreads();   // buffer ib free for the TDM issued next iteration
    }

    // ---- softmax over k (512) : 16 threads per row, width-16 shuffles ----
    {
        const int r = tid >> 4;
        const int t = tid & 15;
        float mx = -3.4e38f;
        for (int kk = t; kk < Lc; kk += 16) mx = fmaxf(mx, sS[r][kk]);
        #pragma unroll
        for (int off = 8; off; off >>= 1) mx = fmaxf(mx, __shfl_xor(mx, off, 16));
        float sum = 0.f;
        for (int kk = t; kk < Lc; kk += 16) {
            float e = __expf(sS[r][kk] - mx);
            sS[r][kk] = e;
            sum += e;
        }
        #pragma unroll
        for (int off = 8; off; off >>= 1) sum += __shfl_xor(sum, off, 16);
        if (t == 0) sInv[r] = 1.f / sum;
    }
    __syncthreads();

    // ---- out = softmax(scores) @ V : 8 waves x one 16-wide d tile each ----
    const int lane = tid & 31;
    const int wave = tid >> 5;
    const int half = lane >> 4;
    const int lr   = lane & 15;
    const int n    = wave * 16 + lr;
    const float* __restrict__ vcol = Vp + (size_t)bL * Dc + n;   // column n of V

    v8f c = {};
    #pragma unroll 4
    for (int k0 = 0; k0 < Lc; k0 += 32) {
        v16bf a, bb;
        const int ka = k0 + half * 8;
        #pragma unroll
        for (int i = 0; i < 8; ++i) {
            a[i]     = (__bf16)sS[lr][ka + i];        // A: M = lr (q row), K split per half
            a[8 + i] = (__bf16)sS[lr][ka + 16 + i];
        }
        const int kb = k0 + half * 8;
        #pragma unroll
        for (int i = 0; i < 8; ++i) {
            bb[i]     = (__bf16)vcol[(size_t)(kb + i) * Dc];   // B[k][n] = V[k, n]
            bb[8 + i] = (__bf16)vcol[(size_t)(kb + 16 + i) * Dc];
        }
        c = __builtin_amdgcn_wmma_f32_16x16x32_bf16(
                false, a, false, bb, (short)0, c, false, false);
    }

    #pragma unroll
    for (int r = 0; r < 8; ++r) {
        const int qrow = half * 8 + r;
        out[(size_t)(bx * 16 + qrow) * Dc + n] = c[r] * sInv[qrow];
    }
}

// ----------------------------------------------------------------------------
extern "C" void kernel_launch(void* const* d_in, const int* in_sizes, int n_in,
                              void* d_out, int out_size, void* d_ws, size_t ws_size,
                              hipStream_t stream) {
    const float* query   = (const float*)d_in[0];
    const float* key     = (const float*)d_in[1];
    const float* value   = (const float*)d_in[2];
    const float* Wq      = (const float*)d_in[3];
    const float* bq      = (const float*)d_in[4];
    const float* Wk      = (const float*)d_in[5];
    const float* bk      = (const float*)d_in[6];
    const float* Wv      = (const float*)d_in[7];
    const float* bv      = (const float*)d_in[8];
    const float* w_delta = (const float*)d_in[9];
    const float* b_delta = (const float*)d_in[10];
    const float* w_sigma = (const float*)d_in[11];
    const float* b_sigma = (const float*)d_in[12];
    const float* w_theta = (const float*)d_in[13];
    const float* b_theta = (const float*)d_in[14];

    float* Qp = (float*)d_ws;            // needs 3*BLD*4 = 6 MB of workspace
    float* Kp = Qp + BLD;
    float* Vp = Kp + BLD;
    float* out = (float*)d_out;

    dim3 pgrid(Bc * Lc / 16, 3);
    proj3_kernel<<<pgrid, 256, 0, stream>>>(query, key, value,
                                            Wq, bq, Wk, bk, Wv, bv,
                                            Qp, Kp, Vp);

    attn_fused_kernel<<<Bc * Lc / 16, 256, 0, stream>>>(
        Qp, Kp, Vp, w_delta, b_delta, w_sigma, b_sigma, w_theta, b_theta, out);
}
